// VN_LocalOri_Globa6D_44478681317790
// MI455X (gfx1250) — compile-verified, matrix operand order checked by reading, not run on hardware
//
#include <hip/hip_runtime.h>

// ---------------------------------------------------------------------------
// VN-DGCNN-style network for gfx1250 (MI455X).
// GEMMs: LDS-staged WMMA (f16 in, f32 acc) with coalesced b128 global loads,
// transposed f16 B panel in LDS so fragments are vector ds_load reads.
// ---------------------------------------------------------------------------

#define EPSV   1e-6f
#define SLOPEV 0.2f

typedef __attribute__((ext_vector_type(16))) _Float16 v16h;
typedef __attribute__((ext_vector_type(8)))  _Float16 v8h;
typedef __attribute__((ext_vector_type(8)))  float    v8f;
typedef __attribute__((ext_vector_type(4)))  float    v4f;

// ---------------------------------------------------------------------------
// Batched GEMM: C[b] = op(A[b]) (MxK) * op(B[b]) (KxN) (+bias per column).
// Block = 128 threads (4 waves). Tile = 16(M) x 64(N), wave w owns cols
// [w*16, w*16+16). K loop in steps of 32 with zero padding.
// A panel staged in LDS as Ash[row][k] (f16), B panel staged transposed as
// BshT[col][k] (f16) so each lane's WMMA fragment is two contiguous 8-half
// runs (vector LDS reads). Fragment layout per ISA 16-bit A table.
// ---------------------------------------------------------------------------
template <int TA, int TB>
__global__ void __launch_bounds__(128)
gemm_wmma_f16_t(const float* __restrict__ A,
                const float* __restrict__ Bm,
                const float* __restrict__ bias,
                float* __restrict__ C,
                int M, int N, int Kd,
                long long sA, long long sB, long long sC,
                int lda, int ldb, int ldc)
{
    __shared__ __align__(16) _Float16 Ash[16 * 32];   // [row][k]
    __shared__ __align__(16) _Float16 BshT[64 * 32];  // [col][k]

    int ntn  = (N + 63) >> 6;
    int tile = blockIdx.x;
    int mt   = tile / ntn;
    int ng   = tile % ntn;
    int b    = blockIdx.y;

    const float* Ab = A  + (long long)b * sA;
    const float* Bb = Bm + (long long)b * sB;
    float*       Cb = C  + (long long)b * sC;

    int tid  = threadIdx.x;        // 0..127
    int wave = tid >> 5;           // 0..3
    int lane = tid & 31;
    int half = lane >> 4;          // 0|1
    int l16  = lane & 15;
    int kb   = half * 8;           // K base within each 16-group

    // cooperative-load index maps
    int a_row    = tid >> 3;       // 0..15
    int a_k      = (tid & 7) * 4;  // 0,4,..,28
    int b_colseg = (tid & 15) * 4; // 0,4,..,60
    int b_krow   = tid >> 4;       // 0..7 (x4 j-steps -> k 0..31)

    v8f acc = {};

    for (int k0 = 0; k0 < Kd; k0 += 32) {
        // ---- stage A panel: 4 clamped scalar loads/thread -----------------
        {
            int grow  = mt * 16 + a_row;
            int growc = grow < M - 1 ? grow : M - 1;
            bool rok  = grow < M;
#pragma unroll
            for (int j = 0; j < 4; ++j) {
                int k  = k0 + a_k + j;
                int kc = k < Kd - 1 ? k : Kd - 1;
                long long ai = TA ? ((long long)kc * lda + growc)
                                  : ((long long)growc * lda + kc);
                float av = Ab[ai];
                Ash[a_row * 32 + a_k + j] =
                    (_Float16)((k < Kd && rok) ? av : 0.f);
            }
        }
        // ---- stage B panel transposed into LDS ----------------------------
        bool bfull = (!TB) && (k0 + 32 <= Kd) && (ng * 64 + 64 <= N) &&
                     ((ldb & 3) == 0);
        if (bfull) {
            // coalesced float4 loads: 128 threads x 4 iters cover 32x64 panel
#pragma unroll
            for (int j = 0; j < 4; ++j) {
                int k   = b_krow + j * 8;
                int col = ng * 64 + b_colseg;
                const v4f* src =
                    (const v4f*)(Bb + (long long)(k0 + k) * ldb + col);
                v4f v = *src;
#pragma unroll
                for (int q = 0; q < 4; ++q)
                    BshT[(b_colseg + q) * 32 + k] = (_Float16)v[q];
            }
        } else {
#pragma unroll
            for (int j = 0; j < 4; ++j) {
                int k  = k0 + b_krow + j * 8;
                int kc = k < Kd - 1 ? k : Kd - 1;
#pragma unroll
                for (int q = 0; q < 4; ++q) {
                    int col  = ng * 64 + b_colseg + q;
                    int colc = col < N - 1 ? col : N - 1;
                    bool ok  = (k < Kd) && (col < N);
                    long long bi = TB ? ((long long)colc * ldb + kc)
                                      : ((long long)kc * ldb + colc);
                    float bv = Bb[bi];
                    BshT[(b_colseg + q) * 32 + (b_krow + j * 8)] =
                        (_Float16)(ok ? bv : 0.f);
                }
            }
        }
        __syncthreads();

        // ---- fragments from LDS: 2x v8h vector reads each -----------------
        const v8h* arow = (const v8h*)(Ash + l16 * 32);
        v8h a0 = arow[kb >> 3];
        v8h a1 = arow[(16 + kb) >> 3];
        const v8h* bcol = (const v8h*)(BshT + (wave * 16 + l16) * 32);
        v8h b0 = bcol[kb >> 3];
        v8h b1 = bcol[(16 + kb) >> 3];
        v16h af, bf;
#pragma unroll
        for (int e = 0; e < 8; ++e) {
            af[e] = a0[e]; af[8 + e] = a1[e];
            bf[e] = b0[e]; bf[8 + e] = b1[e];
        }
        acc = __builtin_amdgcn_wmma_f32_16x16x32_f16(
                  false, af, false, bf, (short)0, acc, false, false);

        // ---- prefetch next K panel (global_prefetch_b8) -------------------
        if (k0 + 32 < Kd) {
            long long ai = TA ? ((long long)(k0 + 32) * lda + mt * 16)
                              : ((long long)(mt * 16) * lda + k0 + 32);
            __builtin_prefetch(&Ab[ai], 0, 1);
            long long bi = TB ? ((long long)(ng * 64) * ldb + k0 + 32)
                              : ((long long)(k0 + 32) * ldb + ng * 64 + b_colseg);
            __builtin_prefetch(&Bb[bi], 0, 1);
        }
        __syncthreads();
    }

    // ---- epilogue ---------------------------------------------------------
    int ccol = ng * 64 + wave * 16 + l16;
#pragma unroll
    for (int r = 0; r < 8; ++r) {
        int rr = mt * 16 + r + half * 8;
        if (rr < M && ccol < N) {
            float v = acc[r];
            if (bias) v += bias[ccol];
            Cb[(long long)rr * ldc + ccol] = v;
        }
    }
}

// ---------------------------------------------------------------------------
// sq[b][n] = sum_c X[b][c][n]^2     (X: B x Cf x N)
// ---------------------------------------------------------------------------
__global__ void sqnorm_kernel(const float* __restrict__ X, float* __restrict__ sq,
                              int Cf, int N)
{
    int b = blockIdx.y;
    int n = blockIdx.x * blockDim.x + threadIdx.x;
    if (n >= N) return;
    const float* Xb = X + (long long)b * Cf * N;
    float s = 0.f;
    for (int c = 0; c < Cf; ++c) { float v = Xb[(long long)c * N + n]; s += v * v; }
    sq[b * N + n] = s;
}

// ---------------------------------------------------------------------------
// Per-row top-k of dist = 2*inner - sq[n] - sq[m]. One block per (b,n) row.
// Iterative argmax extraction, lowest-index tie-break (matches lax.top_k).
// ---------------------------------------------------------------------------
#define TKT 256
__global__ void topk_kernel(const float* __restrict__ inner,
                            const float* __restrict__ sq,
                            int* __restrict__ idx, int N, int Kk)
{
    __shared__ float vals[2048];
    __shared__ float rv[TKT];
    __shared__ int   ri[TKT];
    int b = blockIdx.y, n = blockIdx.x;
    const float* row = inner + ((long long)b * N + n) * N;
    float sqn = sq[b * N + n];
    for (int m = threadIdx.x; m < N; m += TKT)
        vals[m] = 2.f * row[m] - sqn - sq[b * N + m];
    __syncthreads();
    for (int t = 0; t < Kk; ++t) {
        float best = -3.4e38f; int bi = 0x7fffffff;
        for (int m = threadIdx.x; m < N; m += TKT) {
            float v = vals[m];
            if (v > best || (v == best && m < bi)) { best = v; bi = m; }
        }
        rv[threadIdx.x] = best; ri[threadIdx.x] = bi;
        __syncthreads();
        for (int s = TKT / 2; s > 0; s >>= 1) {
            if (threadIdx.x < s) {
                float ov = rv[threadIdx.x + s]; int oi = ri[threadIdx.x + s];
                if (ov > rv[threadIdx.x] ||
                    (ov == rv[threadIdx.x] && oi < ri[threadIdx.x])) {
                    rv[threadIdx.x] = ov; ri[threadIdx.x] = oi;
                }
            }
            __syncthreads();
        }
        if (threadIdx.x == 0) {
            idx[((long long)b * N + n) * Kk + t] = ri[0];
            vals[ri[0]] = -3.4e38f;
        }
        __syncthreads();
    }
}

// ---------------------------------------------------------------------------
// g[b, c, d, n, k]: c<C -> x[b,c,d,idx]-x[b,c,d,n] ; c>=C -> x[b,c-C,d,n]
// ---------------------------------------------------------------------------
__global__ void graph_feature_kernel(const float* __restrict__ x,
                                     const int* __restrict__ idx,
                                     float* __restrict__ g,
                                     int Cd, int N, int Kk)
{
    int b = blockIdx.y;
    long long per = 2LL * Cd * 3 * N * Kk;
    long long t = (long long)blockIdx.x * blockDim.x + threadIdx.x;
    if (t >= per) return;
    int k = (int)(t % Kk); long long r = t / Kk;
    int n = (int)(r % N);  r /= N;
    int d = (int)(r % 3);  r /= 3;
    int c = (int)r;
    const float* xb = x + (long long)b * Cd * 3 * N;
    float ctr = xb[((long long)(c % Cd) * 3 + d) * N + n];
    float outv;
    if (c < Cd) {
        int j = idx[((long long)b * N + n) * Kk + k];
        outv = xb[((long long)c * 3 + d) * N + j] - ctr;
    } else {
        outv = ctr;
    }
    g[(long long)b * per + t] = outv;
}

// ---------------------------------------------------------------------------
// VN batch-norm statistics: per-channel sum & sumsq of n = |p|+eps.
// p laid out (B, C, 3, S). grid.y = channel.
// ---------------------------------------------------------------------------
__global__ void bn_stats_kernel(const float* __restrict__ p,
                                float* __restrict__ stats,
                                int Cd, int B, long long S)
{
    __shared__ float s1[256], s2[256];
    int c = blockIdx.y;
    long long BS = (long long)B * S;
    float a1 = 0.f, a2 = 0.f;
    for (long long t = (long long)blockIdx.x * blockDim.x + threadIdx.x;
         t < BS; t += (long long)gridDim.x * blockDim.x) {
        long long b = t / S, s = t % S;
        const float* pp = p + (((long long)b * Cd + c) * 3) * S + s;
        float x0 = pp[0], x1 = pp[S], x2 = pp[2 * S];
        float nn = sqrtf(x0 * x0 + x1 * x1 + x2 * x2) + EPSV;
        a1 += nn; a2 += nn * nn;
    }
    s1[threadIdx.x] = a1; s2[threadIdx.x] = a2;
    __syncthreads();
    for (int s = 128; s > 0; s >>= 1) {
        if (threadIdx.x < s) {
            s1[threadIdx.x] += s1[threadIdx.x + s];
            s2[threadIdx.x] += s2[threadIdx.x + s];
        }
        __syncthreads();
    }
    if (threadIdx.x == 0) {
        atomicAdd(&stats[2 * c],     s1[0]);
        atomicAdd(&stats[2 * c + 1], s2[0]);
    }
}

// ---------------------------------------------------------------------------
// VN nonlinearity: BN-normalize p, directional leaky-ReLU against d.
// ---------------------------------------------------------------------------
__global__ void vn_combine_kernel(const float* __restrict__ p,
                                  const float* __restrict__ dmat,
                                  const float* __restrict__ stats,
                                  float* __restrict__ out,
                                  int Cd, int CdD, int B, long long S)
{
    long long tot = (long long)B * Cd * S;
    long long t = (long long)blockIdx.x * blockDim.x + threadIdx.x;
    if (t >= tot) return;
    long long s = t % S; long long r = t / S;
    int c = (int)(r % Cd); int b = (int)(r / Cd);
    const float* pp = p + (((long long)b * Cd + c) * 3) * S + s;
    int cd = (CdD == 1) ? 0 : c;
    const float* dd = dmat + (((long long)b * CdD + cd) * 3) * S + s;
    float cnt = (float)B * (float)S;
    float m   = stats[2 * c] / cnt;
    float var = stats[2 * c + 1] / cnt - m * m;
    float p0 = pp[0], p1 = pp[S], p2 = pp[2 * S];
    float n  = sqrtf(p0 * p0 + p1 * p1 + p2 * p2) + EPSV;
    float nb = (n - m) * rsqrtf(var + 1e-5f);
    float sc = nb / n;
    p0 *= sc; p1 *= sc; p2 *= sc;
    float d0 = dd[0], d1 = dd[S], d2 = dd[2 * S];
    float dot = p0 * d0 + p1 * d1 + p2 * d2;
    float dsq = d0 * d0 + d1 * d1 + d2 * d2;
    float f   = dot / (dsq + EPSV);
    float q0 = (dot >= 0.f) ? p0 : p0 - f * d0;
    float q1 = (dot >= 0.f) ? p1 : p1 - f * d1;
    float q2 = (dot >= 0.f) ? p2 : p2 - f * d2;
    float* oo = out + (((long long)b * Cd + c) * 3) * S + s;
    oo[0]     = SLOPEV * p0 + (1.f - SLOPEV) * q0;
    oo[S]     = SLOPEV * p1 + (1.f - SLOPEV) * q1;
    oo[2 * S] = SLOPEV * p2 + (1.f - SLOPEV) * q2;
}

// ---------------------------------------------------------------------------
// Mean over innermost dim.
// ---------------------------------------------------------------------------
__global__ void mean_lastdim_kernel(const float* __restrict__ in,
                                    float* __restrict__ out,
                                    long long outer, int Kk)
{
    long long t = (long long)blockIdx.x * blockDim.x + threadIdx.x;
    if (t >= outer) return;
    const float* p = in + t * Kk;
    float s = 0.f;
    for (int k = 0; k < Kk; ++k) s += p[k];
    out[t] = s / (float)Kk;
}

// ---------------------------------------------------------------------------
// cat = [x ; cross(x, d)]   (B, 2C, 3, N)
// ---------------------------------------------------------------------------
__global__ void cross_concat_kernel(const float* __restrict__ x,
                                    const float* __restrict__ d,
                                    float* __restrict__ cat,
                                    int Cd, int N, int B)
{
    long long tot = (long long)B * Cd * N;
    long long t = (long long)blockIdx.x * blockDim.x + threadIdx.x;
    if (t >= tot) return;
    int n = (int)(t % N); long long r = t / N;
    int c = (int)(r % Cd); int b = (int)(r / Cd);
    const float* xp = x + (((long long)b * Cd + c) * 3) * N + n;
    const float* dp = d + (((long long)b * Cd + c) * 3) * N + n;
    float x0 = xp[0], x1 = xp[N], x2 = xp[2 * N];
    float d0 = dp[0], d1 = dp[N], d2 = dp[2 * N];
    float c0 = x1 * d2 - x2 * d1;
    float c1 = x2 * d0 - x0 * d2;
    float c2 = x0 * d1 - x1 * d0;
    float* o1 = cat + (((long long)b * 2 * Cd + c) * 3) * N + n;
    float* o2 = cat + (((long long)b * 2 * Cd + Cd + c) * 3) * N + n;
    o1[0] = x0; o1[N] = x1; o1[2 * N] = x2;
    o2[0] = c0; o2[N] = c1; o2[2 * N] = c2;
}

// ---------------------------------------------------------------------------
// Copy channels: src (B,Csrc,S3) -> dst (B,Cdst,S3) at channel offset.
// ---------------------------------------------------------------------------
__global__ void copy_channels_kernel(const float* __restrict__ src,
                                     float* __restrict__ dst,
                                     int Csrc, int Cdst, int coff,
                                     long long S3, int B)
{
    long long tot = (long long)B * Csrc * S3;
    long long t = (long long)blockIdx.x * blockDim.x + threadIdx.x;
    if (t >= tot) return;
    long long s = t % S3; long long r = t / S3;
    int c = (int)(r % Csrc); int b = (int)(r / Csrc);
    dst[((long long)b * Cdst + coff + c) * S3 + s] =
        src[((long long)b * Csrc + c) * S3 + s];
}

// ---------------------------------------------------------------------------
// affine_geometric: Gram-Schmidt frame (a1,a2) -> project v.
// v,a1,a2,y: (B, F, 3) contiguous 3-vectors.
// ---------------------------------------------------------------------------
__global__ void affine_kernel(const float* __restrict__ v,
                              const float* __restrict__ a1,
                              const float* __restrict__ a2,
                              float* __restrict__ y, int F, int B)
{
    long long tot = (long long)B * F;
    long long t = (long long)blockIdx.x * blockDim.x + threadIdx.x;
    if (t >= tot) return;
    const float* vp = v  + t * 3;
    const float* p1 = a1 + t * 3;
    const float* p2 = a2 + t * 3;
    float a10 = p1[0], a11 = p1[1], a12 = p1[2];
    float n1 = sqrtf(a10 * a10 + a11 * a11 + a12 * a12) + EPSV;
    float e10 = a10 / n1, e11 = a11 / n1, e12 = a12 / n1;
    float a20 = p2[0], a21 = p2[1], a22 = p2[2];
    float pr = e10 * a20 + e11 * a21 + e12 * a22;
    float u0 = a20 - pr * e10, u1 = a21 - pr * e11, u2 = a22 - pr * e12;
    float n2 = sqrtf(u0 * u0 + u1 * u1 + u2 * u2) + EPSV;
    float e20 = u0 / n2, e21 = u1 / n2, e22 = u2 / n2;
    float e30 = e11 * e22 - e12 * e21;
    float e31 = e12 * e20 - e10 * e22;
    float e32 = e10 * e21 - e11 * e20;
    float v0 = vp[0], v1 = vp[1], v2 = vp[2];
    float* yp = y + t * 3;
    yp[0] = v0 * e10 + v1 * e11 + v2 * e12;
    yp[1] = v0 * e20 + v1 * e21 + v2 * e22;
    yp[2] = v0 * e30 + v1 * e31 + v2 * e32;
}

__global__ void zero_kernel(float* __restrict__ p, long long n)
{
    long long t = (long long)blockIdx.x * blockDim.x + threadIdx.x;
    if (t < n) p[t] = 0.f;
}

// ===========================================================================
// Host orchestration
// ===========================================================================
static inline long long cdivll(long long a, long long b) { return (a + b - 1) / b; }

static void gemm(hipStream_t st, const float* A, const float* Bm, float* C,
                 int M, int Nc, int Kd, int batch,
                 long long sA, long long sB, long long sC,
                 int lda, int ldb, int ldc, int tA, int tB, const float* bias)
{
    dim3 grid((unsigned)(cdivll(M, 16) * cdivll(Nc, 64)), (unsigned)batch);
    if (tA)
        gemm_wmma_f16_t<1, 0><<<grid, 128, 0, st>>>(A, Bm, bias, C, M, Nc, Kd, sA, sB, sC, lda, ldb, ldc);
    else if (tB)
        gemm_wmma_f16_t<0, 1><<<grid, 128, 0, st>>>(A, Bm, bias, C, M, Nc, Kd, sA, sB, sC, lda, ldb, ldc);
    else
        gemm_wmma_f16_t<0, 0><<<grid, 128, 0, st>>>(A, Bm, bias, C, M, Nc, Kd, sA, sB, sC, lda, ldb, ldc);
}

static void vn_block_run(hipStream_t st, const float* xin,
                         const float* W, const float* U,
                         int Cin, int Cout, int CU, int B, long long S,
                         float* ptmp, float* dtmp, float* stats, float* out)
{
    int N3 = (int)(3 * S);
    gemm(st, W, xin, ptmp, Cout, N3, Cin, B,
         0, (long long)Cin * N3, (long long)Cout * N3, Cin, N3, N3, 0, 0, nullptr);
    gemm(st, U, xin, dtmp, CU, N3, Cin, B,
         0, (long long)Cin * N3, (long long)CU * N3, Cin, N3, N3, 0, 0, nullptr);
    zero_kernel<<<(unsigned)cdivll(2 * Cout, 256), 256, 0, st>>>(stats, 2 * Cout);
    long long chunks = cdivll((long long)B * S, 256);
    if (chunks > 512) chunks = 512;
    bn_stats_kernel<<<dim3((unsigned)chunks, (unsigned)Cout), 256, 0, st>>>(
        ptmp, stats, Cout, B, S);
    long long tot = (long long)B * Cout * S;
    vn_combine_kernel<<<(unsigned)cdivll(tot, 256), 256, 0, st>>>(
        ptmp, dtmp, stats, out, Cout, CU, B, S);
}

static void local_geo_run(hipStream_t st, const float* xin,
                          const float* A, const float* Bw,
                          int C, int B, int N, float* dtmp, float* cat, float* out)
{
    int N3 = 3 * N;
    gemm(st, A, xin, dtmp, C, N3, C, B,
         0, (long long)C * N3, (long long)C * N3, C, N3, N3, 0, 0, nullptr);
    long long tot = (long long)B * C * N;
    cross_concat_kernel<<<(unsigned)cdivll(tot, 256), 256, 0, st>>>(xin, dtmp, cat, C, N, B);
    gemm(st, Bw, cat, out, C, N3, 2 * C, B,
         0, (long long)(2 * C) * N3, (long long)C * N3, 2 * C, N3, N3, 0, 0, nullptr);
}

static void knn_run(hipStream_t st, const float* flat, int Cf, int B, int N, int Kk,
                    float* inner, float* sq, int* idx)
{
    sqnorm_kernel<<<dim3((unsigned)cdivll(N, 256), (unsigned)B), 256, 0, st>>>(flat, sq, Cf, N);
    gemm(st, flat, flat, inner, N, N, Cf, B,
         (long long)Cf * N, (long long)Cf * N, (long long)N * N,
         N, N, N, /*transA=*/1, 0, nullptr);
    topk_kernel<<<dim3((unsigned)N, (unsigned)B), 256, 0, st>>>(inner, sq, idx, N, Kk);
}

extern "C" void kernel_launch(void* const* d_in, const int* in_sizes, int n_in,
                              void* d_out, int out_size, void* d_ws, size_t ws_size,
                              hipStream_t stream)
{
    (void)in_sizes; (void)n_in; (void)out_size; (void)ws_size;
    const float* x       = (const float*)d_in[0];
    const float* conv1_W = (const float*)d_in[1];
    const float* conv1_U = (const float*)d_in[2];
    const float* conv2_W = (const float*)d_in[3];
    const float* conv2_U = (const float*)d_in[4];
    const float* lg1_A   = (const float*)d_in[5];
    const float* lg1_B   = (const float*)d_in[6];
    const float* conv3_W = (const float*)d_in[7];
    const float* conv3_U = (const float*)d_in[8];
    const float* conv4_W = (const float*)d_in[9];
    const float* conv4_U = (const float*)d_in[10];
    const float* lg2_A   = (const float*)d_in[11];
    const float* lg2_B   = (const float*)d_in[12];
    const float* conv5_W = (const float*)d_in[13];
    const float* conv5_U = (const float*)d_in[14];
    const float* conv6_W = (const float*)d_in[15];
    const float* conv6_U = (const float*)d_in[16];
    const float* lin1    = (const float*)d_in[17];
    const float* lin2    = (const float*)d_in[18];
    const float* lin2_   = (const float*)d_in[19];
    const float* lin3    = (const float*)d_in[20];
    const float* lin4    = (const float*)d_in[21];
    const float* lin4_   = (const float*)d_in[22];
    const float* inv1_W  = (const float*)d_in[23];
    const float* inv1_U  = (const float*)d_in[24];
    const float* inv2_W  = (const float*)d_in[25];
    const float* inv2_U  = (const float*)d_in[26];
    const float* inv3_W  = (const float*)d_in[27];
    const float* lin0_W  = (const float*)d_in[28];
    const float* lin0_b  = (const float*)d_in[29];

    const int B = 4, N = 2048, Kk = 20, C0 = 21, FEAT = 512;
    const long long Snk = (long long)N * Kk;

    // ---- workspace bump allocator -----------------------------------------
    char* wsp = (char*)d_ws;
    size_t off = 0;
    auto allocf = [&](long long elems) -> float* {
        off = (off + 255) & ~(size_t)255;
        float* p = (float*)(wsp + off);
        off += (size_t)elems * sizeof(float);
        return p;
    };
    auto alloci = [&](long long elems) -> int* {
        off = (off + 255) & ~(size_t)255;
        int* p = (int*)(wsp + off);
        off += (size_t)elems * sizeof(int);
        return p;
    };

    float* inner  = allocf((long long)B * N * N);              // 64 MB
    float* sq     = allocf((long long)B * N);
    int*   idx    = alloci((long long)B * N * Kk);
    float* gbuf   = allocf((long long)B * 2 * C0 * 3 * Snk);   // 42ch graph feat
    float* pbuf   = allocf((long long)B * C0 * 3 * Snk);
    float* dbuf   = allocf((long long)B * C0 * 3 * Snk);
    float* hbuf   = allocf((long long)B * C0 * 3 * Snk);
    float* hmean  = allocf((long long)B * C0 * 3 * N);
    float* dgeo   = allocf((long long)B * C0 * 3 * N);
    float* catbuf = allocf((long long)B * 2 * C0 * 3 * N);
    float* x1     = allocf((long long)B * C0 * 3 * N);
    float* x2     = allocf((long long)B * C0 * 3 * N);
    float* x3     = allocf((long long)B * C0 * 3 * N);
    float* x123   = allocf((long long)B * 3 * C0 * 3 * N);
    float* p6     = allocf((long long)B * FEAT * 3 * N);
    float* d6     = allocf((long long)B * 1 * 3 * N);
    float* h6     = allocf((long long)B * FEAT * 3 * N);
    float* hmean6 = allocf((long long)B * FEAT * 3);
    float* hfeat  = allocf((long long)B * 2 * FEAT * 3);
    float* vbuf   = allocf((long long)B * FEAT * 3);
    float* jbuf   = allocf((long long)B * FEAT * 3);
    float* jpbuf  = allocf((long long)B * FEAT * 3);
    float* ybuf   = allocf((long long)B * FEAT * 3);
    float* comb   = allocf((long long)B * 2 * FEAT * 3);
    float* x2b    = allocf((long long)B * FEAT * 3);
    float* j2     = allocf((long long)B * FEAT * 3);
    float* j2p    = allocf((long long)B * FEAT * 3);
    float* y2     = allocf((long long)B * FEAT * 3);
    float* outbuf = allocf((long long)B * 2 * FEAT * 3);
    float* z1p    = allocf((long long)B * FEAT * 3);
    float* z1d    = allocf((long long)B * FEAT * 3);
    float* z1     = allocf((long long)B * FEAT * 3);
    float* z2p    = allocf((long long)B * (FEAT / 2) * 3);
    float* z2d    = allocf((long long)B * (FEAT / 2) * 3);
    float* z2     = allocf((long long)B * (FEAT / 2) * 3);
    float* z0     = allocf((long long)B * 9);
    float* xinv   = allocf((long long)B * 2 * FEAT * 3);
    float* stats  = allocf(2 * FEAT);

    // ---- stage 1: graph on raw points (C=1, flat dim 3) -------------------
    knn_run(stream, x, 3, B, N, Kk, inner, sq, idx);
    {
        long long per = 2LL * 1 * 3 * N * Kk;
        graph_feature_kernel<<<dim3((unsigned)cdivll(per, 256), B), 256, 0, stream>>>(
            x, idx, gbuf, 1, N, Kk);
    }
    vn_block_run(stream, gbuf, conv1_W, conv1_U, 2, C0, C0, B, Snk, pbuf, dbuf, stats, hbuf);
    vn_block_run(stream, hbuf, conv2_W, conv2_U, C0, C0, C0, B, Snk, pbuf, dbuf, stats, hbuf);
    mean_lastdim_kernel<<<(unsigned)cdivll((long long)B * C0 * 3 * N, 256), 256, 0, stream>>>(
        hbuf, hmean, (long long)B * C0 * 3 * N, Kk);
    local_geo_run(stream, hmean, lg1_A, lg1_B, C0, B, N, dgeo, catbuf, x1);

    // ---- stage 2 ----------------------------------------------------------
    knn_run(stream, x1, 3 * C0, B, N, Kk, inner, sq, idx);
    {
        long long per = 2LL * C0 * 3 * N * Kk;
        graph_feature_kernel<<<dim3((unsigned)cdivll(per, 256), B), 256, 0, stream>>>(
            x1, idx, gbuf, C0, N, Kk);
    }
    vn_block_run(stream, gbuf, conv3_W, conv3_U, 2 * C0, C0, C0, B, Snk, pbuf, dbuf, stats, hbuf);
    vn_block_run(stream, hbuf, conv4_W, conv4_U, C0, C0, C0, B, Snk, pbuf, dbuf, stats, hbuf);
    mean_lastdim_kernel<<<(unsigned)cdivll((long long)B * C0 * 3 * N, 256), 256, 0, stream>>>(
        hbuf, hmean, (long long)B * C0 * 3 * N, Kk);
    local_geo_run(stream, hmean, lg2_A, lg2_B, C0, B, N, dgeo, catbuf, x2);

    // ---- stage 3 ----------------------------------------------------------
    knn_run(stream, x2, 3 * C0, B, N, Kk, inner, sq, idx);
    {
        long long per = 2LL * C0 * 3 * N * Kk;
        graph_feature_kernel<<<dim3((unsigned)cdivll(per, 256), B), 256, 0, stream>>>(
            x2, idx, gbuf, C0, N, Kk);
    }
    vn_block_run(stream, gbuf, conv5_W, conv5_U, 2 * C0, C0, C0, B, Snk, pbuf, dbuf, stats, hbuf);
    mean_lastdim_kernel<<<(unsigned)cdivll((long long)B * C0 * 3 * N, 256), 256, 0, stream>>>(
        hbuf, hmean, (long long)B * C0 * 3 * N, Kk);
    local_geo_run(stream, hmean, lg2_A, lg2_B, C0, B, N, dgeo, catbuf, x3);

    // ---- conv6 on concat(x1,x2,x3) ----------------------------------------
    long long S3N = 3LL * N;
    copy_channels_kernel<<<(unsigned)cdivll((long long)B * C0 * S3N, 256), 256, 0, stream>>>(
        x1, x123, C0, 3 * C0, 0, S3N, B);
    copy_channels_kernel<<<(unsigned)cdivll((long long)B * C0 * S3N, 256), 256, 0, stream>>>(
        x2, x123, C0, 3 * C0, C0, S3N, B);
    copy_channels_kernel<<<(unsigned)cdivll((long long)B * C0 * S3N, 256), 256, 0, stream>>>(
        x3, x123, C0, 3 * C0, 2 * C0, S3N, B);
    vn_block_run(stream, x123, conv6_W, conv6_U, 3 * C0, FEAT, 1, B, N, p6, d6, stats, h6);

    // global mean over N, duplicate to 1024 channels
    mean_lastdim_kernel<<<(unsigned)cdivll((long long)B * FEAT * 3, 256), 256, 0, stream>>>(
        h6, hmean6, (long long)B * FEAT * 3, N);
    copy_channels_kernel<<<(unsigned)cdivll((long long)B * FEAT * 3, 256), 256, 0, stream>>>(
        hmean6, hfeat, FEAT, 2 * FEAT, 0, 3, B);
    copy_channels_kernel<<<(unsigned)cdivll((long long)B * FEAT * 3, 256), 256, 0, stream>>>(
        hmean6, hfeat, FEAT, 2 * FEAT, FEAT, 3, B);

    // ---- head: equivariant frame ------------------------------------------
    gemm(stream, lin1,  hfeat, vbuf,  FEAT, 3, 2 * FEAT, B, 0, 2LL * FEAT * 3, (long long)FEAT * 3, 2 * FEAT, 3, 3, 0, 0, nullptr);
    gemm(stream, lin2,  hfeat, jbuf,  FEAT, 3, 2 * FEAT, B, 0, 2LL * FEAT * 3, (long long)FEAT * 3, 2 * FEAT, 3, 3, 0, 0, nullptr);
    gemm(stream, lin2_, hfeat, jpbuf, FEAT, 3, 2 * FEAT, B, 0, 2LL * FEAT * 3, (long long)FEAT * 3, 2 * FEAT, 3, 3, 0, 0, nullptr);
    affine_kernel<<<(unsigned)cdivll((long long)B * FEAT, 256), 256, 0, stream>>>(
        vbuf, jbuf, jpbuf, ybuf, FEAT, B);
    copy_channels_kernel<<<(unsigned)cdivll((long long)B * FEAT * 3, 256), 256, 0, stream>>>(
        vbuf, comb, FEAT, 2 * FEAT, 0, 3, B);
    copy_channels_kernel<<<(unsigned)cdivll((long long)B * FEAT * 3, 256), 256, 0, stream>>>(
        ybuf, comb, FEAT, 2 * FEAT, FEAT, 3, B);
    gemm(stream, lin3,  comb, x2b, FEAT, 3, 2 * FEAT, B, 0, 2LL * FEAT * 3, (long long)FEAT * 3, 2 * FEAT, 3, 3, 0, 0, nullptr);
    gemm(stream, lin4,  comb, j2,  FEAT, 3, 2 * FEAT, B, 0, 2LL * FEAT * 3, (long long)FEAT * 3, 2 * FEAT, 3, 3, 0, 0, nullptr);
    gemm(stream, lin4_, comb, j2p, FEAT, 3, 2 * FEAT, B, 0, 2LL * FEAT * 3, (long long)FEAT * 3, 2 * FEAT, 3, 3, 0, 0, nullptr);
    affine_kernel<<<(unsigned)cdivll((long long)B * FEAT, 256), 256, 0, stream>>>(
        x2b, j2, j2p, y2, FEAT, B);
    copy_channels_kernel<<<(unsigned)cdivll((long long)B * FEAT * 3, 256), 256, 0, stream>>>(
        x2b, outbuf, FEAT, 2 * FEAT, 0, 3, B);
    copy_channels_kernel<<<(unsigned)cdivll((long long)B * FEAT * 3, 256), 256, 0, stream>>>(
        y2, outbuf, FEAT, 2 * FEAT, FEAT, 3, B);

    // first output: `out` (B, 1024, 3)
    hipMemcpyAsync(d_out, outbuf, (size_t)B * 2 * FEAT * 3 * sizeof(float),
                   hipMemcpyDeviceToDevice, stream);

    // ---- invariant branch ---------------------------------------------------
    vn_block_run(stream, outbuf, inv1_W, inv1_U, 2 * FEAT, FEAT, FEAT, B, 1, z1p, z1d, stats, z1);
    vn_block_run(stream, z1, inv2_W, inv2_U, FEAT, FEAT / 2, FEAT / 2, B, 1, z2p, z2d, stats, z2);
    gemm(stream, inv3_W, z2, z0, 3, 3, FEAT / 2, B,
         0, (long long)(FEAT / 2) * 3, 9, FEAT / 2, 3, 3, 0, 0, nullptr);
    // x_inv = out @ z0    (1024x3)(3x3)
    gemm(stream, outbuf, z0, xinv, 2 * FEAT, 3, 3, B,
         2LL * FEAT * 3, 9, 2LL * FEAT * 3, 3, 3, 3, 0, 0, nullptr);
    // x_out1 = x_inv @ lin0_W^T + b   (1024x3)(3x1024)
    gemm(stream, xinv, lin0_W, (float*)d_out + (long long)B * 2 * FEAT * 3,
         2 * FEAT, 1024, 3, B,
         2LL * FEAT * 3, 0, 1024LL * 1024, 3, 3, 1024, 0, /*transB=*/1, lin0_b);
}
